// SAM_77653008712343
// MI455X (gfx1250) — matrix operand order repaired and checked by reading, not compiled, not run on hardware
//
#include <hip/hip_runtime.h>
#include <hip/hip_bf16.h>

// ---------------------------------------------------------------------------
// Problem constants (from reference): B=4, C=512, H=W=64, N=4096, IC=256
// ---------------------------------------------------------------------------
#define BB 4
#define CC 512
#define ICC 256
#define NN 4096

typedef __bf16 bf16_t;
typedef __bf16 v16bf __attribute__((ext_vector_type(16)));
typedef float  v8f   __attribute__((ext_vector_type(8)));

union FragAB { v16bf bf; uint4 u[2]; };

__device__ inline v8f v8zero() {
    v8f z = {0.f,0.f,0.f,0.f,0.f,0.f,0.f,0.f};
    return z;
}

__device__ inline v8f wmma_bf16(const FragAB& a, const FragAB& b, v8f c) {
    return __builtin_amdgcn_wmma_f32_16x16x32_bf16(
        false, a.bf, false, b.bf, (short)0, c, false, false);
}

// gfx1250 async global->LDS copy (ASYNCcnt-tracked), GVS mode:
//   lds[lds_addr] = mem[saddr + voff] (16 bytes per lane)
__device__ inline void async_copy_b128(unsigned int lds_addr, unsigned int voff,
                                       const void* sbase) {
    asm volatile("global_load_async_to_lds_b128 %0, %1, %2"
                 :: "v"(lds_addr), "v"(voff), "s"(sbase) : "memory");
}
__device__ inline void wait_async0() {
    asm volatile("s_wait_asynccnt 0x0" ::: "memory");
}

// ---------------------------------------------------------------------------
// Kernel 0: convert weights fp32 -> bf16 (row-major, K-contiguous)
// ---------------------------------------------------------------------------
__global__ __launch_bounds__(256) void k_prepack(
    const float* __restrict__ qw, const float* __restrict__ kw,
    const float* __restrict__ vw,
    bf16_t* __restrict__ qwb, bf16_t* __restrict__ kwb,
    bf16_t* __restrict__ vwb)
{
    int i = blockIdx.x * 256 + threadIdx.x;
    if (i < ICC * CC) { qwb[i] = (bf16_t)qw[i]; kwb[i] = (bf16_t)kw[i]; }
    if (i < CC * CC)  { vwb[i] = (bf16_t)vw[i]; }
}

// ---------------------------------------------------------------------------
// Kernel 1: projection for Q or K.
//   P[m,ic] = relu(W@F + b)[ic,m] + relu(cw@coord + cb)[ic,m]
//   Stores Pt bf16 [B][N][IC] (transposed) and psum[B][N] = sum_ic P^2
//   GEMM: M=16 positions (A from LDS-transposed features), N=ic, K=C.
// ---------------------------------------------------------------------------
__global__ __launch_bounds__(128) void k_proj_qk(
    const float* __restrict__ F,     // [B][C][N]
    const bf16_t* __restrict__ Wb,   // [IC][C] bf16
    const float* __restrict__ bias,  // [IC]
    const float* __restrict__ cw,    // [IC][2]
    const float* __restrict__ cb,    // [IC]
    bf16_t* __restrict__ Pt,         // [B][N][IC]
    float* __restrict__ psum)        // [B][N]
{
    const int bz   = blockIdx.y;
    const int m0   = blockIdx.x * 64;
    const int tid  = threadIdx.x;
    const int w    = tid >> 5;
    const int lane = tid & 31;
    const int lm   = lane & 15;
    const int half = lane >> 4;

    __shared__ __align__(16) bf16_t Ft[64 * 32];   // [pos][c-chunk], c contiguous

    v8f acc[16];
#pragma unroll
    for (int t = 0; t < 16; ++t) acc[t] = v8zero();

    for (int kc = 0; kc < 16; ++kc) {
        // cooperative transpose-load of features chunk: 32 channels x 64 pos
#pragma unroll
        for (int i = 0; i < 16; ++i) {
            int idx = tid + i * 128;          // 0..2047
            int cl  = idx >> 6;               // 0..31
            int ml  = idx & 63;               // 0..63
            const float* src = &F[((size_t)bz * CC + kc * 32 + cl) * NN + m0 + ml];
            Ft[ml * 32 + cl] = (bf16_t)(*src);
            if (((i & 3) == 0) && kc < 15)
                __builtin_prefetch(src + 32 * NN, 0, 1);   // next c-chunk
        }
        __syncthreads();

        FragAB a;   // A: rows = positions (this wave's 16), K = c
        const bf16_t* arow = &Ft[(w * 16 + lm) * 32];
        a.u[0] = *(const uint4*)(arow + 16 * half);
        a.u[1] = *(const uint4*)(arow + 16 * half + 8);

#pragma unroll
        for (int t = 0; t < 16; ++t) {
            FragAB b;   // B: cols = ic, K = c contiguous from prepacked weights
            const bf16_t* brow = &Wb[(size_t)(t * 16 + lm) * CC + kc * 32 + 16 * half];
            b.u[0] = *(const uint4*)(brow);
            b.u[1] = *(const uint4*)(brow + 8);
            acc[t] = wmma_bf16(a, b, acc[t]);
        }
        __syncthreads();
    }

    // epilogue: bias + relu + coord-conv + bf16 round + sum of squares
    float sq[8];
#pragma unroll
    for (int g = 0; g < 8; ++g) sq[g] = 0.f;

#pragma unroll
    for (int t = 0; t < 16; ++t) {
        int oc = t * 16 + lm;
        float bv  = bias[oc];
        float c0  = cw[oc * 2 + 0];
        float c1  = cw[oc * 2 + 1];
        float cbv = cb[oc];
#pragma unroll
        for (int g = 0; g < 8; ++g) {
            int m  = m0 + w * 16 + g + 8 * half;
            int wx = m & 63;
            int hy = (m >> 6) & 63;
            float lw = -1.f + (float)wx * (2.f / 63.f);
            float lh = -1.f + (float)hy * (2.f / 63.f);
            float ccv = fmaxf(0.f, c0 * lw + c1 * lh + cbv);
            float v   = fmaxf(0.f, acc[t][g] + bv) + ccv;
            bf16_t vb = (bf16_t)v;
            float vf  = (float)vb;               // use bf16-rounded value
            sq[g] += vf * vf;
            Pt[((size_t)bz * NN + m) * ICC + oc] = vb;
        }
    }
#pragma unroll
    for (int g = 0; g < 8; ++g) {
        float s = sq[g];
        s += __shfl_xor(s, 1);
        s += __shfl_xor(s, 2);
        s += __shfl_xor(s, 4);
        s += __shfl_xor(s, 8);
        if (lm == 0)
            psum[(size_t)bz * NN + m0 + w * 16 + g + 8 * half] = s;
    }
}

// ---------------------------------------------------------------------------
// Kernel 2: V projection.  V[c,m] = relu(vw@F + vb).  Stores bf16 [B][C][N].
//   GEMM: M=c (A from prepacked vw), N=16 positions (B from LDS features).
// ---------------------------------------------------------------------------
__global__ __launch_bounds__(128) void k_proj_v(
    const float* __restrict__ F,     // [B][C][N]
    const bf16_t* __restrict__ Vw,   // [C][C] bf16
    const float* __restrict__ vb,    // [C]
    bf16_t* __restrict__ Vbf)        // [B][C][N]
{
    const int bz   = blockIdx.y;
    const int m0   = blockIdx.x * 16;
    const int tid  = threadIdx.x;
    const int w    = tid >> 5;
    const int lane = tid & 31;
    const int lm   = lane & 15;
    const int half = lane >> 4;

    __shared__ __align__(16) bf16_t Ft[16 * 32];   // [pos][c-chunk]

    v8f acc[8];
#pragma unroll
    for (int j = 0; j < 8; ++j) acc[j] = v8zero();

    for (int kc = 0; kc < 16; ++kc) {
#pragma unroll
        for (int i = 0; i < 4; ++i) {
            int idx = tid + i * 128;          // 0..511
            int cl  = idx >> 4;               // 0..31
            int ml  = idx & 15;               // 0..15
            const float* src = &F[((size_t)bz * CC + kc * 32 + cl) * NN + m0 + ml];
            Ft[ml * 32 + cl] = (bf16_t)(*src);
            if ((i == 0) && kc < 15)
                __builtin_prefetch(src + 32 * NN, 0, 1);
        }
        __syncthreads();

        FragAB b;   // B: cols = positions, K = c contiguous
        const bf16_t* brow = &Ft[lm * 32 + 16 * half];
        b.u[0] = *(const uint4*)(brow);
        b.u[1] = *(const uint4*)(brow + 8);

#pragma unroll
        for (int j = 0; j < 8; ++j) {
            int c = (w * 8 + j) * 16 + lm;
            FragAB a;
            const bf16_t* ar = &Vw[(size_t)c * CC + kc * 32];
            a.u[0] = *(const uint4*)(ar + 16 * half);
            a.u[1] = *(const uint4*)(ar + 16 * half + 8);
            acc[j] = wmma_bf16(a, b, acc[j]);
        }
        __syncthreads();
    }

#pragma unroll
    for (int j = 0; j < 8; ++j) {
#pragma unroll
        for (int g = 0; g < 8; ++g) {
            int c = (w * 8 + j) * 16 + g + 8 * half;
            float val = fmaxf(0.f, acc[j][g] + vb[c]);
            Vbf[((size_t)bz * CC + c) * NN + m0 + lm] = (bf16_t)val;
        }
    }
}

// ---------------------------------------------------------------------------
// Kernel 3: rowsum[n] = sum_m exp(-(qs[n]+ks[m]-2*q.k)); stores 1/(rs+1e-14).
//   Per wave: 16 rows, sweep all 4096 m via WMMA QtxKt (K=IC=256).
// ---------------------------------------------------------------------------
__global__ __launch_bounds__(128) void k_rowsum(
    const bf16_t* __restrict__ Qt,   // [B][N][IC]
    const bf16_t* __restrict__ Kt,   // [B][N][IC]
    const float* __restrict__ qs,    // [B][N]
    const float* __restrict__ ks,    // [B][N]
    float* __restrict__ invr)        // [B][N]
{
    const int bz   = blockIdx.y;
    const int tid  = threadIdx.x;
    const int w    = tid >> 5;
    const int lane = tid & 31;
    const int lm   = lane & 15;
    const int half = lane >> 4;
    const int n0   = blockIdx.x * 64 + w * 16;

    FragAB aQ[8];
    const bf16_t* qrow = &Qt[((size_t)bz * NN + n0 + lm) * ICC];
#pragma unroll
    for (int kc = 0; kc < 8; ++kc) {
        aQ[kc].u[0] = *(const uint4*)(qrow + kc * 32 + 16 * half);
        aQ[kc].u[1] = *(const uint4*)(qrow + kc * 32 + 16 * half + 8);
    }
    float qsr[8];
#pragma unroll
    for (int g = 0; g < 8; ++g)
        qsr[g] = qs[(size_t)bz * NN + n0 + g + 8 * half];

    float rs[8];
#pragma unroll
    for (int g = 0; g < 8; ++g) rs[g] = 0.f;

    for (int mt = 0; mt < 256; ++mt) {
        const bf16_t* krow = &Kt[((size_t)bz * NN + mt * 16 + lm) * ICC];
        if (mt + 8 < 256)
            __builtin_prefetch(krow + 8 * 16 * ICC, 0, 1);
        v8f e = v8zero();
#pragma unroll
        for (int kc = 0; kc < 8; ++kc) {
            FragAB bK;
            bK.u[0] = *(const uint4*)(krow + kc * 32 + 16 * half);
            bK.u[1] = *(const uint4*)(krow + kc * 32 + 16 * half + 8);
            e = wmma_bf16(aQ[kc], bK, e);
        }
        float ksm = ks[(size_t)bz * NN + mt * 16 + lm];
#pragma unroll
        for (int g = 0; g < 8; ++g)
            rs[g] += __expf(-(qsr[g] + ksm - 2.f * e[g]));
    }
#pragma unroll
    for (int g = 0; g < 8; ++g) {
        float s = rs[g];
        s += __shfl_xor(s, 1);
        s += __shfl_xor(s, 2);
        s += __shfl_xor(s, 4);
        s += __shfl_xor(s, 8);
        if (lm == 0)
            invr[(size_t)bz * NN + n0 + g + 8 * half] = 1.f / (s + 1e-14f);
    }
}

// ---------------------------------------------------------------------------
// Kernel 4 (fused): out[c,m] = sum_n V[c,n]*exp(-D[n,m])*inv[n] + features.
//   Workgroup = 8 waves, 64 output columns (4 m-tiles), all 512 channels.
//   Per 64-wide n-chunk:
//     * Qt rows for the chunk are staged into LDS ONCE via async global->LDS
//       (ASYNCcnt), issued so the copy for chunk i+1 overlaps the V*E' WMMAs
//       of chunk i.
//     * each wave regenerates two 16x16 E' tiles (exp(-D)*inv) via WMMA+exp,
//       parks them bf16 in LDS ([m][n], n-contiguous = B layout),
//     * all waves run the V x E' accumulation WMMAs.
// ---------------------------------------------------------------------------
__global__ __launch_bounds__(256) void k_attn_out(
    const bf16_t* __restrict__ Qt,   // [B][N][IC]
    const bf16_t* __restrict__ Kt,   // [B][N][IC]
    const bf16_t* __restrict__ Vbf,  // [B][C][N]
    const float* __restrict__ qs,    // [B][N]
    const float* __restrict__ ks,    // [B][N]
    const float* __restrict__ invr,  // [B][N]
    const float* __restrict__ F,     // [B][C][N] residual
    float* __restrict__ out)         // [B][C][N]
{
    const int bz   = blockIdx.y;
    const int m0   = blockIdx.x * 64;
    const int tid  = threadIdx.x;
    const int w    = tid >> 5;        // 0..7
    const int lane = tid & 31;
    const int lm   = lane & 15;
    const int half = lane >> 4;
    const int mt_e = w & 3;           // this wave's E m-tile (fixed)
    const int nsb  = w >> 2;          // base n-sub (0/1); tiles nsb and nsb+2

    __shared__ __align__(16) bf16_t QtS[64 * 256];   // 32 KB staged Qt chunk
    __shared__ __align__(16) bf16_t El[64 * 64];     // 8 KB E' chunk [m][n]

    // resident Kt B-fragments for this wave's E column tile
    FragAB bK[8];
    const bf16_t* krow = &Kt[((size_t)bz * NN + m0 + mt_e * 16 + lm) * ICC];
#pragma unroll
    for (int kc = 0; kc < 8; ++kc) {
        bK[kc].u[0] = *(const uint4*)(krow + kc * 32 + 16 * half);
        bK[kc].u[1] = *(const uint4*)(krow + kc * 32 + 16 * half + 8);
    }
    const float ksm = ks[(size_t)bz * NN + m0 + mt_e * 16 + lm];

    v8f accO[16];
#pragma unroll
    for (int i = 0; i < 16; ++i) accO[i] = v8zero();

    const unsigned int lds_qts = (unsigned int)(uintptr_t)(&QtS[0]);

    // stage chunk 0 (32 KB: 256 threads x 8 x 16B)
    {
        const bf16_t* qblk = &Qt[(size_t)bz * NN * ICC];
#pragma unroll
        for (int i = 0; i < 8; ++i) {
            unsigned int off = (unsigned int)(tid + i * 256) * 16u;
            async_copy_b128(lds_qts + off, off, qblk);
        }
        wait_async0();
    }
    __syncthreads();

    for (int it = 0; it < 64; ++it) {
        const int nb = it * 64;

        // --- E' tiles: exp(-D)*inv, bf16, into LDS (Qt read from LDS stage) ---
#pragma unroll
        for (int t2 = 0; t2 < 2; ++t2) {
            const int ns = nsb + 2 * t2;
            const bf16_t* qrow = &QtS[(ns * 16 + lm) * ICC];
            v8f e = v8zero();
#pragma unroll
            for (int kc = 0; kc < 8; ++kc) {
                FragAB a;
                a.u[0] = *(const uint4*)(qrow + kc * 32 + 16 * half);
                a.u[1] = *(const uint4*)(qrow + kc * 32 + 16 * half + 8);
                e = wmma_bf16(a, bK[kc], e);
            }
#pragma unroll
            for (int g = 0; g < 8; ++g) {
                int nr = nb + ns * 16 + g + 8 * half;
                float qsr = qs[(size_t)bz * NN + nr];
                float iv  = invr[(size_t)bz * NN + nr];
                float ep  = __expf(-(qsr + ksm - 2.f * e[g])) * iv;
                El[(mt_e * 16 + lm) * 64 + ns * 16 + g + 8 * half] = (bf16_t)ep;
            }
        }
        __syncthreads();   // QtS reads + El writes complete

        // --- async-stage next Qt chunk; overlaps the out-phase WMMAs ---
        if (it + 1 < 64) {
            const bf16_t* qblk = &Qt[((size_t)bz * NN + (it + 1) * 64) * ICC];
#pragma unroll
            for (int i = 0; i < 8; ++i) {
                unsigned int off = (unsigned int)(tid + i * 256) * 16u;
                async_copy_b128(lds_qts + off, off, qblk);
            }
        }

        // --- accumulate out += V[:, nb..nb+63] x E'[nb..nb+63, :] ---
#pragma unroll
        for (int ksub = 0; ksub < 2; ++ksub) {
#pragma unroll
            for (int j = 0; j < 4; ++j) {
                int c = (w * 4 + j) * 16 + lm;
                FragAB aV;
                const bf16_t* vrow = &Vbf[((size_t)bz * CC + c) * NN + nb + ksub * 32];
                aV.u[0] = *(const uint4*)(vrow + 16 * half);
                aV.u[1] = *(const uint4*)(vrow + 16 * half + 8);
#pragma unroll
                for (int mt = 0; mt < 4; ++mt) {
                    FragAB bE;
                    const bf16_t* erow = &El[(mt * 16 + lm) * 64 + ksub * 32 + 16 * half];
                    bE.u[0] = *(const uint4*)(erow);
                    bE.u[1] = *(const uint4*)(erow + 8);
                    accO[j * 4 + mt] = wmma_bf16(aV, bE, accO[j * 4 + mt]);
                }
            }
        }
        wait_async0();     // next QtS chunk landed
        __syncthreads();   // El reads complete; safe to overwrite both buffers
    }

    // epilogue: residual add, fp32 store
#pragma unroll
    for (int j = 0; j < 4; ++j) {
#pragma unroll
        for (int mt = 0; mt < 4; ++mt) {
#pragma unroll
            for (int g = 0; g < 8; ++g) {
                int c = (w * 4 + j) * 16 + g + 8 * half;
                int m = m0 + mt * 16 + lm;
                size_t idx = ((size_t)bz * CC + c) * NN + m;
                out[idx] = accO[j * 4 + mt][g] + F[idx];
            }
        }
    }
}

// ---------------------------------------------------------------------------
// Launch
// ---------------------------------------------------------------------------
extern "C" void kernel_launch(void* const* d_in, const int* in_sizes, int n_in,
                              void* d_out, int out_size, void* d_ws, size_t ws_size,
                              hipStream_t stream) {
    const float* F  = (const float*)d_in[0];
    const float* qw = (const float*)d_in[1];
    const float* qb = (const float*)d_in[2];
    const float* kw = (const float*)d_in[3];
    const float* kb = (const float*)d_in[4];
    const float* vw = (const float*)d_in[5];
    const float* vb = (const float*)d_in[6];
    const float* cw = (const float*)d_in[7];
    const float* cb = (const float*)d_in[8];

    char* ws = (char*)d_ws;
    size_t off = 0;
    bf16_t* qwb = (bf16_t*)(ws + off); off += (size_t)ICC * CC * 2;      // 256 KB
    bf16_t* kwb = (bf16_t*)(ws + off); off += (size_t)ICC * CC * 2;      // 256 KB
    bf16_t* vwb = (bf16_t*)(ws + off); off += (size_t)CC * CC * 2;       // 512 KB
    bf16_t* Qt  = (bf16_t*)(ws + off); off += (size_t)BB * NN * ICC * 2; // 8 MB
    bf16_t* Kt  = (bf16_t*)(ws + off); off += (size_t)BB * NN * ICC * 2; // 8 MB
    bf16_t* Vb  = (bf16_t*)(ws + off); off += (size_t)BB * CC * NN * 2;  // 16 MB
    float* qsb  = (float*)(ws + off);  off += (size_t)BB * NN * 4;
    float* ksb  = (float*)(ws + off);  off += (size_t)BB * NN * 4;
    float* ivb  = (float*)(ws + off);  off += (size_t)BB * NN * 4;

    k_prepack<<<dim3((CC * CC) / 256), 256, 0, stream>>>(qw, kw, vw, qwb, kwb, vwb);
    k_proj_qk<<<dim3(NN / 64, BB), 128, 0, stream>>>(F, qwb, qb, cw, cb, Qt, qsb);
    k_proj_qk<<<dim3(NN / 64, BB), 128, 0, stream>>>(F, kwb, kb, cw, cb, Kt, ksb);
    k_proj_v <<<dim3(NN / 16, BB), 128, 0, stream>>>(F, vwb, vb, Vb);
    k_rowsum <<<dim3(NN / 64, BB), 128, 0, stream>>>(Qt, Kt, qsb, ksb, ivb);
    k_attn_out<<<dim3(NN / 64, BB), 256, 0, stream>>>(Qt, Kt, Vb, qsb, ksb, ivb, F,
                                                      (float*)d_out);
}
